// ModConv_74165495267655
// MI455X (gfx1250) — matrix operand order repaired and checked by reading, compiled, not run on hardware
//
#include <hip/hip_runtime.h>
#include <hip/hip_bf16.h>

// Problem constants (from reference)
#define B_SZ   16
#define CIN    256
#define COUT   256
#define SDIM   512
#define HH     64
#define WW     64
#define KG     2304   // CIN * 3 * 3
#define KC     32     // K-chunk per WMMA sweep
#define NKC    72     // KG / KC  (K reordered as k' = rs*256 + cin)
#define MT     256    // M tile per block (all of COUT)
#define NT     128    // N tile per block
#define NPIX   4096   // HH*WW

typedef __bf16 bf16;
typedef bf16  v8bf  __attribute__((ext_vector_type(8)));
typedef bf16  v16bf __attribute__((ext_vector_type(16)));
typedef float v4f   __attribute__((ext_vector_type(4)));
typedef float v8f   __attribute__((ext_vector_type(8)));

// ---------------------------------------------------------------------------
// style[b][cin] = dot(w[b,:], mod_w[cin,:]) + mod_b[cin]
// ---------------------------------------------------------------------------
__global__ __launch_bounds__(256)
void style_kernel(const float* __restrict__ w, const float* __restrict__ mod_w,
                  const float* __restrict__ mod_b, float* __restrict__ style) {
    int idx = blockIdx.x * 256 + threadIdx.x;      // 0 .. B*CIN-1
    int b   = idx >> 8;
    int cin = idx & 255;
    const float* wr = w     + (size_t)b   * SDIM;
    const float* mr = mod_w + (size_t)cin * SDIM;
    float s = 0.f;
    #pragma unroll 8
    for (int j = 0; j < SDIM; ++j) s += wr[j] * mr[j];
    style[idx] = s + mod_b[cin];
}

// ---------------------------------------------------------------------------
// demod[b][cout] = rsqrt( sum_k (weight[cout,k] * style[b, k/9])^2 + eps )
// one wave32 per (b, cout)
// ---------------------------------------------------------------------------
__global__ __launch_bounds__(256)
void demod_kernel(const float* __restrict__ weight, const float* __restrict__ style,
                  float* __restrict__ demod) {
    int lane = threadIdx.x & 31;
    int wvG  = blockIdx.x * 8 + (threadIdx.x >> 5); // 0 .. B*COUT-1
    int b    = wvG >> 8;
    int cout = wvG & 255;
    const float* wr = weight + (size_t)cout * KG;
    const float* st = style  + (size_t)b * CIN;
    float sum = 0.f;
    for (int k = lane; k < KG; k += 32) {
        float v = wr[k] * st[k / 9];
        sum += v * v;
    }
    #pragma unroll
    for (int o = 16; o >= 1; o >>= 1) sum += __shfl_xor(sum, o, 32);
    if (lane == 0) demod[wvG] = rsqrtf(sum + 1e-8f);
}

// ---------------------------------------------------------------------------
// weight transpose: [cout][cin][rs] -> wT[cout][rs][cin]  (fp32, one-time)
// makes the rs-major K ordering load contiguously (b128) in the GEMM fill
// ---------------------------------------------------------------------------
__global__ __launch_bounds__(256)
void wtrans_kernel(const float* __restrict__ weight, float* __restrict__ wT) {
    int idx  = blockIdx.x * 256 + threadIdx.x;   // 0 .. COUT*CIN*9-1
    int rs   = idx % 9;
    int t    = idx / 9;                          // cout*256 + cin
    int cin  = t & 255;
    int cout = t >> 8;
    wT[((size_t)cout * 9 + rs) * CIN + cin] = weight[idx];
}

// ---------------------------------------------------------------------------
// Implicit-GEMM modulated conv, bf16 WMMA with f32 accumulation.
// K reordered as k' = rs*256 + cin -> per 32-chunk: fixed (r,s), contiguous cin.
// B tile (x, fp32) via CDNA5 async global->LDS; A tile = wT * style * demod,
// loaded as b128 from the transposed weights, stored bf16.
// Block: 256 threads (8 waves). Tile: 256(M=cout) x 128(N=pixel),
// wave tile 64x64 = 16 WMMAs per wave per k-step.
// ---------------------------------------------------------------------------
__global__ __launch_bounds__(256)
void modconv_wmma(const float* __restrict__ x, const float* __restrict__ wT,
                  const float* __restrict__ style, const float* __restrict__ demod,
                  float* __restrict__ out) {
    __shared__ __align__(16) bf16  Ald[2][MT][KC];   // scaled weights (bf16) 32 KB
    __shared__ __align__(16) float Bld[2][NT][KC];   // raw x tile (fp32)    32 KB
    __shared__ __align__(16) float styleS[CIN];
    __shared__ __align__(16) float demodS[MT];

    const int tid   = threadIdx.x;
    const int b     = blockIdx.z;
    const int nBase = blockIdx.x * NT;

    styleS[tid] = style[(size_t)b * CIN  + tid];
    demodS[tid] = demod[(size_t)b * COUT + tid];
    __syncthreads();

    // ---- fill geometry ----
    // A: thread owns one cout row (tid), all 32 cin of the chunk
    // B: thread owns one pixel (tid>>1), 16 cin of the chunk
    const int   pFill = tid >> 1;
    const int   iBase = (tid & 1) * 16;
    const float* xb   = x + (size_t)b * CIN * NPIX;
    const float* wRow = wT + (size_t)tid * 9 * CIN;
    const int   p     = nBase + pFill;
    const int   ph    = p >> 6;
    const int   pw    = p & 63;
    const unsigned bldBase = (unsigned)(uintptr_t)&Bld[0][0][0];

    bf16 aReg[32];

    // B tile: pure copy of x -> async global->LDS (fp32, no modulation needed)
    auto issueAsyncB = [&](int kc, int buf) {
        const int rs      = kc >> 3;
        const int cinBase = (kc & 7) * 32;
        const int r  = rs / 3;
        const int s  = rs - r * 3;
        int ih = ph + r - 1; ih = ih < 0 ? -ih : (ih > 63 ? 126 - ih : ih);
        int iw = pw + s - 1; iw = iw < 0 ? -iw : (iw > 63 ? 126 - iw : iw);
        const float* gp = xb + (size_t)(cinBase + iBase) * NPIX + ih * WW + iw;
        unsigned lds = bldBase + (unsigned)(((buf * NT + pFill) * KC + iBase) * 4);
        #pragma unroll
        for (int i = 0; i < 16; ++i) {
            asm volatile("global_load_async_to_lds_b32 %0, %1, off"
                         :: "v"(lds), "v"(gp) : "memory");
            gp  += NPIX;   // next cin plane (+16 KB)
            lds += 4;
        }
    };

    // A tile: wT[cout][rs][cin] * style[cin] * demod[cout] -> bf16 regs
    auto fillA = [&](int kc) {
        const int rs      = kc >> 3;
        const int cinBase = (kc & 7) * 32;
        const float dsc   = demodS[tid];
        const v4f* wp     = (const v4f*)(wRow + (size_t)rs * CIN + cinBase);
        const v4f* sp     = (const v4f*)&styleS[cinBase];
        #pragma unroll
        for (int q = 0; q < 8; ++q) {
            v4f wv4 = wp[q];
            v4f sv4 = sp[q];
            #pragma unroll
            for (int e = 0; e < 4; ++e)
                aReg[q * 4 + e] = (bf16)(wv4[e] * sv4[e] * dsc);
        }
    };
    auto storeA = [&](int buf) {
        #pragma unroll
        for (int q = 0; q < 4; ++q)
            *(v8bf*)&Ald[buf][tid][q * 8] = *(v8bf*)&aReg[q * 8];
    };

    // ---- per-wave WMMA geometry: 8 waves = 4(M) x 2(N), each 64x64 ----
    const int wv   = tid >> 5;
    const int lane = tid & 31;
    const int wM   = (wv & 3) * 64;
    const int wN   = (wv >> 2) * 64;
    const int lm   = lane & 15;
    const int lhi  = lane >> 4;             // 0/1 half-wave
    const int aOff = lhi * 16;              // A-frag k run:  [aOff,+16)
    const int bOff = lhi * 16;              // B-frag k run:  [bOff,+16)

    v8f acc[4][4] = {};

    auto compute = [&](int buf) {
        v16bf afrag[4], bfrag[4];
        #pragma unroll
        for (int t = 0; t < 4; ++t) {
            const bf16* ap = &Ald[buf][wM + t * 16 + lm][0];
            v8bf lo = *(const v8bf*)(ap + aOff);
            v8bf hi = *(const v8bf*)(ap + aOff + 8);
            afrag[t] = __builtin_shufflevector(lo, hi,
                        0,1,2,3,4,5,6,7,8,9,10,11,12,13,14,15);
        }
        #pragma unroll
        for (int u = 0; u < 4; ++u) {
            const float* bp = &Bld[buf][wN + u * 16 + lm][bOff];
            v16bf f;
            #pragma unroll
            for (int e = 0; e < 16; ++e) f[e] = (bf16)bp[e]; // v_cvt_pk_bf16_f32
            bfrag[u] = f;
        }
        #pragma unroll
        for (int t = 0; t < 4; ++t)
            #pragma unroll
            for (int u = 0; u < 4; ++u)
                acc[t][u] = __builtin_amdgcn_wmma_f32_16x16x32_bf16(
                    false, afrag[t], false, bfrag[u],
                    (short)0, acc[t][u], false, false);
    };

    // ---- software-pipelined main loop: one barrier per k-step ----
    issueAsyncB(0, 0);
    fillA(0);
    storeA(0);
    asm volatile("s_wait_asynccnt 0x0" ::: "memory");
    __syncthreads();

    for (int kc = 0; kc < NKC; ++kc) {
        const int cur = kc & 1;
        const int nxt = cur ^ 1;
        if (kc + 1 < NKC) {
            issueAsyncB(kc + 1, nxt);  // buf nxt last read in previous step (fenced)
            fillA(kc + 1);
        }
        compute(cur);
        if (kc + 1 < NKC) {
            storeA(nxt);
            asm volatile("s_wait_asynccnt 0x0" ::: "memory");
            __syncthreads();
        }
    }

    // ---- store D fragments (f32 16x16: lane = n + 16*(m>=8), vgpr = m%8) ----
    float* ob = out + (size_t)b * COUT * NPIX + nBase;
    #pragma unroll
    for (int t = 0; t < 4; ++t) {
        #pragma unroll
        for (int u = 0; u < 4; ++u) {
            int mrow = wM + t * 16 + lhi * 8;
            int ncol = wN + u * 16 + lm;
            #pragma unroll
            for (int v = 0; v < 8; ++v)
                ob[(size_t)(mrow + v) * NPIX + ncol] = acc[t][u][v];
        }
    }
}

// ---------------------------------------------------------------------------
extern "C" void kernel_launch(void* const* d_in, const int* in_sizes, int n_in,
                              void* d_out, int out_size, void* d_ws, size_t ws_size,
                              hipStream_t stream) {
    const float* x      = (const float*)d_in[0];  // [B, CIN, H, W]
    const float* w      = (const float*)d_in[1];  // [B, SDIM]
    const float* weight = (const float*)d_in[2];  // [COUT, CIN, 3, 3]
    const float* mod_w  = (const float*)d_in[3];  // [CIN, SDIM]
    const float* mod_b  = (const float*)d_in[4];  // [CIN]
    float* out = (float*)d_out;

    float* style = (float*)d_ws;                  // [B, CIN]   = 4096 f32
    float* demod = style + (B_SZ * CIN);          // [B, COUT]  = 4096 f32
    float* wT    = demod + (B_SZ * COUT);         // [COUT][9][CIN] = 589824 f32

    style_kernel<<<(B_SZ * CIN) / 256, 256, 0, stream>>>(w, mod_w, mod_b, style);
    demod_kernel<<<(B_SZ * COUT) / 8 / 32, 256, 0, stream>>>(weight, style, demod);
    wtrans_kernel<<<(COUT * CIN * 9) / 256, 256, 0, stream>>>(weight, wT);

    dim3 grid(NPIX / NT, COUT / MT, B_SZ);        // (32, 1, 16)
    modconv_wmma<<<grid, 256, 0, stream>>>(x, wT, style, demod, out);
}
